// Encoder_6442450944673
// MI455X (gfx1250) — matrix operand order repaired and checked by reading, not compiled, not run on hardware
//
#include <hip/hip_runtime.h>
#include <hip/hip_bf16.h>
#include <stdint.h>

// ---------------- problem constants ----------------
#define BATCH  256
#define LSEQ   8192
#define WIN    128
#define STRD   32
#define TSTEPS 253          // start(t) = 32*t for t in [0,253)
#define H1     256          // LSTM1 hidden (HID)
#define H2     128          // LSTM2 hidden (EMB)
#define G1     (4*H1)       // 1024 gate rows, order i,f,g,o
#define G2     (4*H2)       // 512
#define KCAT   384          // concat K: H1+WIN == H2+H1
#define MTILE  16           // batch rows per workgroup

// LDS A1 row: [h1 buf0 (256) | h1 buf1 (256) | x window (128)] + pad
#define A1PAD  648              // 1296B/row: %16==0, 324 dw %64 == 4 -> conflict-free
// LDS A2 row: [h2 buf0 (128) | h2 buf1 (128) | h1(t) (256)] + pad
#define A2PAD  520              // 1040B/row: %16==0, 260 dw %64 == 4

typedef __attribute__((ext_vector_type(16))) __bf16 bf16x16;
typedef __attribute__((ext_vector_type(8)))  float  f32x8;

union FragB16 { bf16x16 v; uint4 q[2]; };

__device__ __forceinline__ unsigned short f32_to_bf16_rne(float f) {
    union { float f; uint32_t u; } x; x.f = f;
    uint32_t u = x.u;
    return (unsigned short)((u + 0x7FFFu + ((u >> 16) & 1u)) >> 16);
}
__device__ __forceinline__ float tanh_fast(float x) {
#if __has_builtin(__builtin_amdgcn_tanhf)
    return __builtin_amdgcn_tanhf(x);        // v_tanh_f32 (CDNA5 TRANS op)
#else
    x = fminf(fmaxf(x, -15.0f), 15.0f);
    float e = __expf(2.0f * x);
    return (e - 1.0f) / (e + 1.0f);
#endif
}
__device__ __forceinline__ float sigm(float x) {    // 1/(1+e^-x) = 0.5*tanh(x/2)+0.5
    return fmaf(tanh_fast(0.5f * x), 0.5f, 0.5f);
}

// B-matrix (32x16 bf16) fragment: lane half 0 -> col n, K k0..k0+15; half 1 -> K k0+16..31.
__device__ __forceinline__ bf16x16 load_bfrag(const unsigned short* __restrict__ w,
                                              int n, int k0, int hi) {
    FragB16 f;
    const uint4* p = (const uint4*)(w + (size_t)n * KCAT + k0 + (hi << 4));
    f.q[0] = p[0];
    f.q[1] = p[1];
    return f.v;
}
// A-matrix (16x32 bf16) fragment from LDS row at column base cb:
// half 0 -> K cb..cb+15; half 1 -> K cb+16..cb+31.
__device__ __forceinline__ bf16x16 load_afrag(const unsigned short* rowp, int cb, int hi) {
    FragB16 f;
    int off = cb + (hi << 4);
    f.q[0] = *(const uint4*)(rowp + off);
    f.q[1] = *(const uint4*)(rowp + off + 8);
    return f.v;
}

// ---------------- prep: f32 -> bf16, concatenated weight layouts ----------------
__global__ void prep_wcat1(const float* __restrict__ Whh, const float* __restrict__ Wih,
                           unsigned short* __restrict__ out) {
    int idx = blockIdx.x * blockDim.x + threadIdx.x;
    if (idx >= G1 * KCAT) return;
    int n = idx / KCAT, k = idx % KCAT;
    float v = (k < H1) ? Whh[n * H1 + k] : Wih[n * WIN + (k - H1)];
    out[idx] = f32_to_bf16_rne(v);
}
__global__ void prep_wcat2(const float* __restrict__ Whh, const float* __restrict__ Wih,
                           unsigned short* __restrict__ out) {
    int idx = blockIdx.x * blockDim.x + threadIdx.x;
    if (idx >= G2 * KCAT) return;
    int n = idx / KCAT, k = idx % KCAT;
    float v = (k < H2) ? Whh[n * H2 + k] : Wih[n * H1 + (k - H2)];
    out[idx] = f32_to_bf16_rne(v);
}
__global__ void prep_xbf(const float* __restrict__ x, unsigned short* __restrict__ out) {
    for (int i = blockIdx.x * blockDim.x + threadIdx.x; i < BATCH * LSEQ;
         i += gridDim.x * blockDim.x)
        out[i] = f32_to_bf16_rne(x[i]);
}

// ---------------- fused 2-layer LSTM, persistent over all 253 steps ----------------
__global__ __launch_bounds__(512) void lstm_fused(
    const unsigned short* __restrict__ Wcat1,      // [G1][KCAT] = [Whh1 | Wih1] bf16
    const unsigned short* __restrict__ Wcat2,      // [G2][KCAT] = [Whh2 | Wih2] bf16
    const unsigned short* __restrict__ xbf,        // [B][L] bf16
    const float* __restrict__ b_ih1, const float* __restrict__ b_hh1,
    const float* __restrict__ b_ih2, const float* __restrict__ b_hh2,
    float* __restrict__ out)                       // [B][H2] f32
{
    __shared__ unsigned short A1s[MTILE][A1PAD];
    __shared__ unsigned short A2s[MTILE][A2PAD];

    const int tid  = threadIdx.x;
    const int lane = tid & 31;
    const int wave = tid >> 5;       // 16 waves
    const int hi   = lane >> 4;      // lane half
    const int lm   = lane & 15;
    const int b0   = blockIdx.x * MTILE;

    // zero initial state (parity-0 buffers suffice, but clear everything once)
    for (int i = tid; i < MTILE * A1PAD; i += 512) A1s[i / A1PAD][i % A1PAD] = 0;
    for (int i = tid; i < MTILE * A2PAD; i += 512) A2s[i / A2PAD][i % A2PAD] = 0;

    const int j1 = wave * 16;              // LSTM1: waves 0..15 own hidden [j1, j1+16)
    const int j2 = (wave & 7) * 16;        // LSTM2: waves 0..7 own hidden [j2, j2+16)

    const float bi1 = b_ih1[0 * H1 + j1 + lm] + b_hh1[0 * H1 + j1 + lm];
    const float bf1 = b_ih1[1 * H1 + j1 + lm] + b_hh1[1 * H1 + j1 + lm];
    const float bg1 = b_ih1[2 * H1 + j1 + lm] + b_hh1[2 * H1 + j1 + lm];
    const float bo1 = b_ih1[3 * H1 + j1 + lm] + b_hh1[3 * H1 + j1 + lm];
    const float bi2 = b_ih2[0 * H2 + j2 + lm] + b_hh2[0 * H2 + j2 + lm];
    const float bf2 = b_ih2[1 * H2 + j2 + lm] + b_hh2[1 * H2 + j2 + lm];
    const float bg2 = b_ih2[2 * H2 + j2 + lm] + b_hh2[2 * H2 + j2 + lm];
    const float bo2 = b_ih2[3 * H2 + j2 + lm] + b_hh2[3 * H2 + j2 + lm];

    f32x8 c1 = {0.f,0.f,0.f,0.f,0.f,0.f,0.f,0.f};   // cell state frags (C/D layout)
    f32x8 c2 = {0.f,0.f,0.f,0.f,0.f,0.f,0.f,0.f};

    __syncthreads();

    for (int t = 0; t < TSTEPS; ++t) {
        // Opaque zero offset, redefined every iteration: blocks LICM from hoisting
        // the 96 weight-fragment loads out of the t-loop (register spill disaster),
        // while keeping the *pointers* untouched so InferAddressSpaces still proves
        // the loads are global (global_load_b128, LOADcnt only — not flat/DScnt).
        int wofs = 0;
        asm volatile("" : "+s"(wofs));
        const unsigned short* w1 = Wcat1 + wofs;
        const unsigned short* w2 = Wcat2 + wofs;

        const int rb1 = (t & 1) * H1;          // h1 read buffer base col
        const int wb1 = H1 - rb1;              // h1 write buffer base col
        const int rb2 = (t & 1) * H2;          // h2 read buffer base col
        const int wb2 = H2 - rb2;

        // ---- stage window x[:, 32t : 32t+128] -> A1 cols [512,640) ----
        {
            int m  = tid >> 5;
            int c4 = (tid & 31) * 4;
            uint2 d = *(const uint2*)(xbf + (size_t)(b0 + m) * LSEQ + t * STRD + c4);
            *(uint2*)&A1s[m][2 * H1 + c4] = d;
        }
        __syncthreads();   // (a) x(t) + h2(t-1)/h1-region writes visible

        // ---- LSTM1 gates: A1 [16 x 384] x Wcat1^T via bf16 WMMA ----
        f32x8 ai = {0.f,0.f,0.f,0.f,0.f,0.f,0.f,0.f};
        f32x8 af = ai, ag = ai, ao = ai;
        {
            const unsigned short* rowp = &A1s[lm][0];
            #pragma unroll
            for (int kc = 0; kc < KCAT / 32; ++kc) {
                const int k0 = kc * 32;
                const int cb = (k0 < H1) ? (rb1 + k0) : (H1 + k0);  // xw at 2*H1
                bf16x16 a  = load_afrag(rowp, cb, hi);
                bf16x16 wi = load_bfrag(w1, 0 * H1 + j1 + lm, k0, hi);
                bf16x16 wf = load_bfrag(w1, 1 * H1 + j1 + lm, k0, hi);
                bf16x16 wg = load_bfrag(w1, 2 * H1 + j1 + lm, k0, hi);
                bf16x16 wo = load_bfrag(w1, 3 * H1 + j1 + lm, k0, hi);
                ai = __builtin_amdgcn_wmma_f32_16x16x32_bf16(false, a, false, wi, (short)0, ai, false, false);
                af = __builtin_amdgcn_wmma_f32_16x16x32_bf16(false, a, false, wf, (short)0, af, false, false);
                ag = __builtin_amdgcn_wmma_f32_16x16x32_bf16(false, a, false, wg, (short)0, ag, false, false);
                ao = __builtin_amdgcn_wmma_f32_16x16x32_bf16(false, a, false, wo, (short)0, ao, false, false);
            }
        }

        // ---- LSTM1 elementwise (register-local) -> h1(t) into opposite buffer ----
        {
            #pragma unroll
            for (int r = 0; r < 8; ++r) {
                float gi = sigm(ai[r] + bi1);
                float gf = sigm(af[r] + bf1);
                float gg = tanh_fast(ag[r] + bg1);
                float go = sigm(ao[r] + bo1);
                float c  = gf * c1[r] + gi * gg;
                c1[r] = c;
                unsigned short hb = f32_to_bf16_rne(go * tanh_fast(c));
                int row = r + (hi << 3);
                A1s[row][wb1 + j1 + lm]    = hb;   // recurrent input for t+1 (other parity)
                A2s[row][2 * H2 + j1 + lm] = hb;   // LSTM2 input h1(t)
            }
        }
        __syncthreads();   // (b) h1(t) visible; all phase-A reads complete

        // ---- LSTM2 gates: A2 [16 x 384] x Wcat2^T (waves 0..7) ----
        f32x8 zi = {0.f,0.f,0.f,0.f,0.f,0.f,0.f,0.f};
        f32x8 zf = zi, zg = zi, zo = zi;
        if (wave < 8) {
            const unsigned short* rowp = &A2s[lm][0];
            #pragma unroll
            for (int kc = 0; kc < KCAT / 32; ++kc) {
                const int k0 = kc * 32;
                const int cb = (k0 < H2) ? (rb2 + k0) : (H2 + k0);  // h1 at 2*H2
                bf16x16 a  = load_afrag(rowp, cb, hi);
                bf16x16 wi = load_bfrag(w2, 0 * H2 + j2 + lm, k0, hi);
                bf16x16 wf = load_bfrag(w2, 1 * H2 + j2 + lm, k0, hi);
                bf16x16 wg = load_bfrag(w2, 2 * H2 + j2 + lm, k0, hi);
                bf16x16 wo = load_bfrag(w2, 3 * H2 + j2 + lm, k0, hi);
                zi = __builtin_amdgcn_wmma_f32_16x16x32_bf16(false, a, false, wi, (short)0, zi, false, false);
                zf = __builtin_amdgcn_wmma_f32_16x16x32_bf16(false, a, false, wf, (short)0, zf, false, false);
                zg = __builtin_amdgcn_wmma_f32_16x16x32_bf16(false, a, false, wg, (short)0, zg, false, false);
                zo = __builtin_amdgcn_wmma_f32_16x16x32_bf16(false, a, false, wo, (short)0, zo, false, false);
            }
            #pragma unroll
            for (int r = 0; r < 8; ++r) {
                float gi = sigm(zi[r] + bi2);
                float gf = sigm(zf[r] + bf2);
                float gg = tanh_fast(zg[r] + bg2);
                float go = sigm(zo[r] + bo2);
                float c  = gf * c2[r] + gi * gg;
                c2[r] = c;
                float h  = go * tanh_fast(c);
                int row = r + (hi << 3);
                A2s[row][wb2 + j2 + lm] = f32_to_bf16_rne(h);   // other parity
                if (t == TSTEPS - 1)
                    out[(size_t)(b0 + row) * H2 + j2 + lm] = h; // final h_n
            }
        }
        // next iteration's barrier (a) orders h2(t)/x(t+1) writes vs. their readers
    }
}

// ---------------- host entry ----------------
extern "C" void kernel_launch(void* const* d_in, const int* in_sizes, int n_in,
                              void* d_out, int out_size, void* d_ws, size_t ws_size,
                              hipStream_t stream) {
    (void)in_sizes; (void)n_in; (void)out_size; (void)ws_size;
    const float* x     = (const float*)d_in[0];
    const float* W_ih1 = (const float*)d_in[1];
    const float* W_hh1 = (const float*)d_in[2];
    const float* b_ih1 = (const float*)d_in[3];
    const float* b_hh1 = (const float*)d_in[4];
    const float* W_ih2 = (const float*)d_in[5];
    const float* W_hh2 = (const float*)d_in[6];
    const float* b_ih2 = (const float*)d_in[7];
    const float* b_hh2 = (const float*)d_in[8];

    unsigned short* Wcat1 = (unsigned short*)d_ws;          // 1024*384 bf16
    unsigned short* Wcat2 = Wcat1 + (size_t)G1 * KCAT;      // 512*384  bf16
    unsigned short* xbf   = Wcat2 + (size_t)G2 * KCAT;      // 256*8192 bf16 (~5.4 MB total)

    prep_wcat1<<<(G1 * KCAT + 255) / 256, 256, 0, stream>>>(W_hh1, W_ih1, Wcat1);
    prep_wcat2<<<(G2 * KCAT + 255) / 256, 256, 0, stream>>>(W_hh2, W_ih2, Wcat2);
    prep_xbf<<<1024, 256, 0, stream>>>(x, xbf);

    lstm_fused<<<BATCH / MTILE, 512, 0, stream>>>(
        Wcat1, Wcat2, xbf, b_ih1, b_hh1, b_ih2, b_hh2, (float*)d_out);
}
